// ViT_1717986918782
// MI455X (gfx1250) — matrix operand order repaired and checked by reading, not compiled, not run on hardware
//
#include <hip/hip_runtime.h>
#include <hip/hip_bf16.h>
#include <math.h>

// ---------------------------------------------------------------------------
// DeiT-style Re-attention ViT forward, CDNA5 (gfx1250), wave32 + WMMA bf16.
//
// Key algebraic optimization (exact, not approximate): the reference computes
//   aw = softmax(qk) @ ones(T,T)  ==> aw == 1 everywhere (softmax rows sum to 1)
// so after reattn_w + head-LN the per-head weight is a CONSTANT cnorm[h] and
//   y[b,h,t,d] = cnorm[h] * sum_t' v[b,h,t,d]     (token independent).
// q/k and the T x T attention matrices are dead code; only the V slice of the
// qkv projection is required and proj runs on a 16 x 768 matrix.
// ---------------------------------------------------------------------------

#define BIMG 16
#define TTOK 256
#define EDIM 768
#define FDIM 3072
#define HHEADS 12
#define DH 64
#define NLAYERS 6
#define NPAR 2
#define MROWS (BIMG * TTOK)       // 4096

typedef __attribute__((ext_vector_type(16))) __bf16 v16bf;
typedef __attribute__((ext_vector_type(2)))  __bf16 v2bf;
typedef __attribute__((ext_vector_type(8)))  float  v8f;

// Native conversions (RNE) -> compiler can use v_cvt_(pk_)bf16_f32 hardware.
__device__ __forceinline__ unsigned short f32_to_bf16(float f) {
  union { __bf16 h; unsigned short u; } x;
  x.h = (__bf16)f;
  return x.u;
}

__device__ __forceinline__ unsigned int pack_bf16x2(float lo, float hi) {
  union { v2bf v; unsigned int u; } c;
  c.v[0] = (__bf16)lo;
  c.v[1] = (__bf16)hi;
  return c.u;
}

// ---------------------------------------------------------------------------
// Tiled GEMM: C(MxN) = A(bf16, MxK row-major) * B(fp32, KxN in a ldb-strided
// matrix, converted to bf16 on the fly) + bias, with selectable epilogue.
// Block tile 128x128, K-step 32, double-buffered LDS. 8 waves; each wave owns
// a 32x64 subtile = 2x4 grid of 16x16 accumulators (v_wmma_f32_16x16x32_bf16).
// ---------------------------------------------------------------------------
#define BM 128
#define BN 128
#define BK 32
#define KPAD 8     // row stride 40 ushorts = 80B -> 16B aligned chunks

enum { EPI_STORE_F32 = 0, EPI_GELU_BF16 = 1, EPI_ACCUM_F32 = 2 };

union FragU { uint4 q[2]; v16bf v; };

__device__ __forceinline__ v16bf load_frag_lds(const unsigned short* base, int o0, int o1) {
  FragU u;
  u.q[0] = *reinterpret_cast<const uint4*>(base + o0);   // 8 bf16 (16B, aligned)
  u.q[1] = *reinterpret_cast<const uint4*>(base + o1);
  return u.v;
}

__global__ __launch_bounds__(256)
void gemm_bf16_wmma(const unsigned short* __restrict__ A,   // bf16 bits, lda = K
                    const float* __restrict__ Bsrc,         // fp32, ldb-strided
                    const float* __restrict__ bias,         // length N
                    void* __restrict__ Out,                 // float* or ushort*, ldo = N
                    int M, int N, int K, int ldb, int epi) {
  __shared__ unsigned short As[2][BM][BK + KPAD];   // 2 x 10 KB
  __shared__ unsigned short Bt[2][BN][BK + KPAD];   // 2 x 10 KB, N-major (transposed)

  const int tid  = threadIdx.x;
  const int lane = tid & 31;
  const int wid  = tid >> 5;
  const int m0   = blockIdx.y * BM;
  const int n0   = blockIdx.x * BN;
  const int wm   = (wid & 3) * 32;     // wave row offset inside block tile
  const int wn   = (wid >> 2) * 64;    // wave col offset inside block tile

  // ---- A copy coords: 2 threads/row, 32B (16 bf16) each, row-granular guard
  const int ar  = tid >> 1;
  const int ah  = (tid & 1) * 16;                  // element offset 0 / 16
  const bool aok = (m0 + ar) < M;
  const unsigned short* agp = A + (size_t)(m0 + ar) * K + ah;

  // ---- B copy coords: 2 K-rows x 8 N-cols per thread, float4 coalesced reads
  const int bkp = tid >> 4;                        // 0..15 -> k = 2*bkp, 2*bkp+1
  const int bng = (tid & 15) * 8;                  // n offset
  const float* bgp = Bsrc + (size_t)(2 * bkp) * ldb + n0 + bng;

  uint4  aReg0 = {}, aReg1 = {};
  float4 b00, b01, b10, b11;

  auto fetch = [&](int kt) {
    if (aok) {
      const uint4* p = reinterpret_cast<const uint4*>(agp + kt);
      aReg0 = p[0]; aReg1 = p[1];
    } else {
      aReg0 = uint4{0, 0, 0, 0}; aReg1 = uint4{0, 0, 0, 0};
    }
    const float* q = bgp + (size_t)kt * ldb;
    b00 = *reinterpret_cast<const float4*>(q);
    b01 = *reinterpret_cast<const float4*>(q + 4);
    b10 = *reinterpret_cast<const float4*>(q + ldb);
    b11 = *reinterpret_cast<const float4*>(q + ldb + 4);
  };
  auto stage = [&](int buf) {
    *reinterpret_cast<uint4*>(&As[buf][ar][ah])     = aReg0;
    *reinterpret_cast<uint4*>(&As[buf][ar][ah + 8]) = aReg1;
    float lo[8] = {b00.x, b00.y, b00.z, b00.w, b01.x, b01.y, b01.z, b01.w};
    float hi[8] = {b10.x, b10.y, b10.z, b10.w, b11.x, b11.y, b11.z, b11.w};
#pragma unroll
    for (int i = 0; i < 8; ++i)
      *reinterpret_cast<unsigned int*>(&Bt[buf][bng + i][2 * bkp]) =
          pack_bf16x2(lo[i], hi[i]);
  };

  v8f acc[2][4] = {};

  const int tiles = K / BK;
  fetch(0);
  stage(0);
  __syncthreads();

  for (int kt = 0; kt < tiles; ++kt) {
    const int cur = kt & 1;
    const bool more = (kt + 1) < tiles;
    if (more) fetch((kt + 1) * BK);                // overlap HBM with WMMA below
    if (kt + 2 < tiles) {                          // keep L2 stream warm
      if (aok) __builtin_prefetch(agp + (kt + 2) * BK, 0, 1);
      __builtin_prefetch(bgp + (size_t)(kt + 2) * BK * ldb, 0, 1);
    }

    // --- fragments per ISA 7.12.2 (16-bit A 16x32; B kept N-major) ---------
    const int arow = wm + (lane & 15);
    const int akb  = (lane & 16) ? 8 : 0;          // lanes 16-31: K 8..15 / 24..31
    v16bf aF[2];
    aF[0] = load_frag_lds(&As[cur][arow][0],      akb, akb + 16);
    aF[1] = load_frag_lds(&As[cur][arow + 16][0], akb, akb + 16);

    const int bcol = wn + (lane & 15);
    const int bkb  = (lane & 16) ? 8 : 0;          // lanes 16-31: K 8..15 / 24..31
    v16bf bF[4];
#pragma unroll
    for (int ni = 0; ni < 4; ++ni)
      bF[ni] = load_frag_lds(&Bt[cur][bcol + ni * 16][0], bkb, bkb + 16);

#pragma unroll
    for (int mi = 0; mi < 2; ++mi)
#pragma unroll
      for (int ni = 0; ni < 4; ++ni)
        acc[mi][ni] = __builtin_amdgcn_wmma_f32_16x16x32_bf16(
            false, aF[mi], false, bF[ni], (short)0, acc[mi][ni], false, false);

    if (more) stage(cur ^ 1);
    __syncthreads();
  }

  // --- epilogue: C/D layout = VGPR i -> row i (+8 for lanes>=16), col lane&15
  const int rbase = m0 + wm + ((lane & 16) ? 8 : 0);
  const int cidx  = lane & 15;
#pragma unroll
  for (int mi = 0; mi < 2; ++mi)
#pragma unroll
    for (int ni = 0; ni < 4; ++ni) {
      int gc = n0 + wn + ni * 16 + cidx;
      float bv = bias[gc];
#pragma unroll
      for (int i = 0; i < 8; ++i) {
        int gr = rbase + mi * 16 + i;
        if (gr < M) {
          float v = acc[mi][ni][i] + bv;
          size_t o = (size_t)gr * N + gc;
          if (epi == EPI_STORE_F32) {
            ((float*)Out)[o] = v;
          } else if (epi == EPI_GELU_BF16) {
            float g = 0.5f * v * (1.0f + erff(v * 0.70710678118654752f));
            ((unsigned short*)Out)[o] = f32_to_bf16(g);
          } else {
            ((float*)Out)[o] += v;
          }
        }
      }
    }
}

// ---------------------------------------------------------------------------
// Row LayerNorm over 768 cols. One block per row, 3 elements/thread.
// Optional: add positional embedding (pos[t*768+c], t = row % 256), write
// fp32 and/or bf16 outputs.
// ---------------------------------------------------------------------------
__device__ __forceinline__ void block_meanvar(float s, float ss, float* red,
                                              float& m, float& rstd) {
#pragma unroll
  for (int off = 16; off; off >>= 1) {
    s  += __shfl_down(s, off);
    ss += __shfl_down(ss, off);
  }
  int lane = threadIdx.x & 31, wid = threadIdx.x >> 5;
  if (lane == 0) { red[wid] = s; red[wid + 8] = ss; }
  __syncthreads();
  if (threadIdx.x == 0) {
    float ts = 0.f, tss = 0.f;
    for (int i = 0; i < 8; ++i) { ts += red[i]; tss += red[i + 8]; }
    float mm = ts * (1.0f / 768.0f);
    red[16] = mm;
    red[17] = rsqrtf(tss * (1.0f / 768.0f) - mm * mm + 1e-5f);
  }
  __syncthreads();
  m = red[16]; rstd = red[17];
}

__global__ __launch_bounds__(256)
void ln_rows_kernel(const float* __restrict__ X,
                    const float* __restrict__ w, const float* __restrict__ b,
                    float* __restrict__ outf, unsigned short* __restrict__ outb,
                    const float* __restrict__ pos) {
  __shared__ float red[18];
  const int row = blockIdx.x;
  const float* xr = X + (size_t)row * EDIM;
  float lx[3], s = 0.f, ss = 0.f;
#pragma unroll
  for (int i = 0; i < 3; ++i) {
    float v = xr[threadIdx.x + i * 256];
    lx[i] = v; s += v; ss += v * v;
  }
  float m, r; block_meanvar(s, ss, red, m, r);
#pragma unroll
  for (int i = 0; i < 3; ++i) {
    int c = threadIdx.x + i * 256;
    float y = (lx[i] - m) * r * w[c] + b[c];
    if (pos)  y += pos[(size_t)(row & (TTOK - 1)) * EDIM + c];
    size_t o = (size_t)row * EDIM + c;
    if (outf) outf[o] = y;
    if (outb) outb[o] = f32_to_bf16(y);
  }
}

// Patch rearrange b c (nh p1)(nw p2) -> (p1 p2 c) fused with patch_ln1 -> bf16.
__global__ __launch_bounds__(256)
void patch_ln1_kernel(const float* __restrict__ x,
                      const float* __restrict__ w, const float* __restrict__ b,
                      unsigned short* __restrict__ out) {
  __shared__ float red[18];
  const int row = blockIdx.x;               // b*256 + t
  const int bi = row >> 8, t = row & 255;
  const int ih = t >> 4, iw = t & 15;
  float lx[3], s = 0.f, ss = 0.f;
#pragma unroll
  for (int i = 0; i < 3; ++i) {
    int d = threadIdx.x + i * 256;          // (p1*16 + p2)*3 + c
    int c = d % 3, pp = d / 3, p2 = pp & 15, p1 = pp >> 4;
    float v = x[(((size_t)bi * 3 + c) * 256 + (ih * 16 + p1)) * 256 + (iw * 16 + p2)];
    lx[i] = v; s += v; ss += v * v;
  }
  float m, r; block_meanvar(s, ss, red, m, r);
#pragma unroll
  for (int i = 0; i < 3; ++i) {
    int c = threadIdx.x + i * 256;
    out[(size_t)row * EDIM + c] = f32_to_bf16((lx[i] - m) * r * w[c] + b[c]);
  }
}

// Sv[b][e] = sum_t v[b][t][e]
__global__ void colsum_kernel(const float* __restrict__ v, float* __restrict__ Sv) {
  int idx = blockIdx.x * blockDim.x + threadIdx.x;
  if (idx >= BIMG * EDIM) return;
  int bi = idx / EDIM, e = idx - bi * EDIM;
  const float* p = v + (size_t)bi * TTOK * EDIM + e;
  float s = 0.f;
  for (int t = 0; t < TTOK; ++t) s += p[(size_t)t * EDIM];
  Sv[idx] = s;
}

// cnorm[g] = LN_h( colsum_h reattn_w[h][g] ) * rn_w + rn_b    (12 values)
__global__ void cnorm_kernel(const float* __restrict__ reattn,
                             const float* __restrict__ rnw,
                             const float* __restrict__ rnb,
                             float* __restrict__ cn) {
  __shared__ float c[HHEADS];
  int g = threadIdx.x;
  if (g < HHEADS) {
    float s = 0.f;
    for (int h = 0; h < HHEADS; ++h) s += reattn[h * HHEADS + g];
    c[g] = s;
  }
  __syncthreads();
  if (g < HHEADS) {
    float m = 0.f;
    for (int k = 0; k < HHEADS; ++k) m += c[k];
    m *= (1.0f / HHEADS);
    float v = 0.f;
    for (int k = 0; k < HHEADS; ++k) { float d = c[k] - m; v += d * d; }
    v *= (1.0f / HHEADS);
    cn[g] = (c[g] - m) * rsqrtf(v + 1e-5f) * rnw[g] + rnb[g];
  }
}

// ymat[b][e] = bf16( cnorm[e/64] * Sv[b][e] )   (16 x 768)
__global__ void ys_kernel(const float* __restrict__ Sv, const float* __restrict__ cn,
                          unsigned short* __restrict__ ymat) {
  int idx = blockIdx.x * blockDim.x + threadIdx.x;
  if (idx >= BIMG * EDIM) return;
  int e = idx % EDIM;
  ymat[idx] = f32_to_bf16(cn[e / DH] * Sv[idx]);
}

// h += ls[e] * s ; s is full (rows x E) or broadcast per image (16 x E)
__global__ void residual_kernel(float* __restrict__ h, const float* __restrict__ ls,
                                const float* __restrict__ s, int broadcast) {
  int idx = blockIdx.x * blockDim.x + threadIdx.x;
  if (idx >= MROWS * EDIM) return;
  int e = idx % EDIM;
  float sv = broadcast ? s[(size_t)((idx / EDIM) / TTOK) * EDIM + e] : s[idx];
  h[idx] += ls[e] * sv;
}

// ---------------------------------------------------------------------------
// Host orchestration.  Inputs flatten setup_inputs() order: x, then params in
// dict insertion order (all fp32).
// ---------------------------------------------------------------------------
extern "C" void kernel_launch(void* const* d_in, const int* in_sizes, int n_in,
                              void* d_out, int out_size, void* d_ws, size_t ws_size,
                              hipStream_t stream) {
  (void)in_sizes; (void)n_in; (void)out_size; (void)ws_size;
  const float* x            = (const float*)d_in[0];
  const float* patch_ln1_w  = (const float*)d_in[1];
  const float* patch_ln1_b  = (const float*)d_in[2];
  const float* patch_w      = (const float*)d_in[3];
  const float* patch_b      = (const float*)d_in[4];
  const float* patch_ln2_w  = (const float*)d_in[5];
  const float* patch_ln2_b  = (const float*)d_in[6];
  const float* pos          = (const float*)d_in[7];
  const float* ln1_w        = (const float*)d_in[8];
  const float* ln1_b        = (const float*)d_in[9];
  const float* ln2_w        = (const float*)d_in[10];
  const float* ln2_b        = (const float*)d_in[11];
  const float* ls1          = (const float*)d_in[12];
  const float* ls2          = (const float*)d_in[13];
  const float* qkv_w        = (const float*)d_in[14];
  const float* qkv_b        = (const float*)d_in[15];
  const float* proj_w       = (const float*)d_in[16];
  const float* proj_b       = (const float*)d_in[17];
  // d_in[18..21] = qn_w/qn_b/kn_w/kn_b: dead code (softmax-ones collapse)
  const float* reattn_w     = (const float*)d_in[22];
  const float* rn_w         = (const float*)d_in[23];
  const float* rn_b         = (const float*)d_in[24];
  const float* mlp_ln_w     = (const float*)d_in[25];
  const float* mlp_ln_b     = (const float*)d_in[26];
  const float* mlp_w1       = (const float*)d_in[27];
  const float* mlp_b1       = (const float*)d_in[28];
  const float* mlp_w2       = (const float*)d_in[29];
  const float* mlp_b2       = (const float*)d_in[30];
  const float* norm_w       = (const float*)d_in[31];
  const float* norm_b       = (const float*)d_in[32];

  // ---- workspace carve ----
  char* p = (char*)d_ws;
  auto take = [&](size_t bytes) -> char* {
    char* r = p; p += (bytes + 255) & ~size_t(255); return r;
  };
  float*          hbuf = (float*)take((size_t)MROWS * EDIM * 4);   // residual stream
  float*          f0   = (float*)take((size_t)MROWS * EDIM * 4);   // patch tmp / v buffer
  float*          x2f  = (float*)take((size_t)MROWS * EDIM * 4);   // ln2 output
  float*          msum = (float*)take((size_t)MROWS * EDIM * 4);   // MLP branch sum
  unsigned short* abf  = (unsigned short*)take((size_t)MROWS * EDIM * 2); // bf16 act
  unsigned short* gbf  = (unsigned short*)take((size_t)MROWS * FDIM * 2); // gelu out
  float*          Sv   = (float*)take((size_t)BIMG * EDIM * 4);
  float*          asum = (float*)take((size_t)BIMG * EDIM * 4);
  unsigned short* ymat = (unsigned short*)take((size_t)BIMG * EDIM * 2);
  float*          cn   = (float*)take(HHEADS * 4);

  auto gemm = [&](const unsigned short* A, const float* Bm, const float* bias,
                  void* Out, int M, int N, int K, int ldb, int epi) {
    dim3 grid(N / BN, (M + BM - 1) / BM);
    gemm_bf16_wmma<<<grid, 256, 0, stream>>>(A, Bm, bias, Out, M, N, K, ldb, epi);
  };

  const int nBE = (BIMG * EDIM + 255) / 256;       // 48 blocks
  const int nME = (MROWS * EDIM + 255) / 256;      // 12288 blocks

  // ---- patch embedding ----
  patch_ln1_kernel<<<MROWS, 256, 0, stream>>>(x, patch_ln1_w, patch_ln1_b, abf);
  gemm(abf, patch_w, patch_b, f0, MROWS, EDIM, EDIM, EDIM, EPI_STORE_F32);
  ln_rows_kernel<<<MROWS, 256, 0, stream>>>(f0, patch_ln2_w, patch_ln2_b,
                                            hbuf, nullptr, pos);

  for (int l = 0; l < NLAYERS; ++l) {
    // ---------------- attention block (collapsed) ----------------
    ln_rows_kernel<<<MROWS, 256, 0, stream>>>(hbuf, ln1_w + l * EDIM, ln1_b + l * EDIM,
                                              nullptr, abf, nullptr);
    hipMemsetAsync(asum, 0, (size_t)BIMG * EDIM * 4, stream);
    for (int j = 0; j < NPAR; ++j) {
      size_t lj = (size_t)(l * NPAR + j);
      // V = x1 @ qkv_w[:, 2E:3E] + qkv_b[2E:3E]
      const float* Wv = qkv_w + lj * EDIM * (3 * EDIM) + 2 * EDIM;
      const float* bv = qkv_b + lj * (3 * EDIM) + 2 * EDIM;
      gemm(abf, Wv, bv, f0, MROWS, EDIM, EDIM, 3 * EDIM, EPI_STORE_F32);
      colsum_kernel<<<nBE, 256, 0, stream>>>(f0, Sv);
      cnorm_kernel<<<1, 32, 0, stream>>>(reattn_w + lj * HHEADS * HHEADS,
                                         rn_w + lj * HHEADS, rn_b + lj * HHEADS, cn);
      ys_kernel<<<nBE, 256, 0, stream>>>(Sv, cn, ymat);
      // proj on the 16 x 768 token-independent matrix, summed over branches
      gemm(ymat, proj_w + lj * EDIM * EDIM, proj_b + lj * EDIM, asum,
           BIMG, EDIM, EDIM, EDIM, EPI_ACCUM_F32);
    }
    residual_kernel<<<nME, 256, 0, stream>>>(hbuf, ls1 + l * EDIM, asum, 1);

    // ---------------- MLP block ----------------
    ln_rows_kernel<<<MROWS, 256, 0, stream>>>(hbuf, ln2_w + l * EDIM, ln2_b + l * EDIM,
                                              x2f, nullptr, nullptr);
    hipMemsetAsync(msum, 0, (size_t)MROWS * EDIM * 4, stream);
    for (int j = 0; j < NPAR; ++j) {
      size_t lj = (size_t)(l * NPAR + j);
      ln_rows_kernel<<<MROWS, 256, 0, stream>>>(x2f, mlp_ln_w + lj * EDIM,
                                                mlp_ln_b + lj * EDIM,
                                                nullptr, abf, nullptr);
      gemm(abf, mlp_w1 + lj * (size_t)EDIM * FDIM, mlp_b1 + lj * FDIM, gbf,
           MROWS, FDIM, EDIM, FDIM, EPI_GELU_BF16);
      gemm(gbf, mlp_w2 + lj * (size_t)FDIM * EDIM, mlp_b2 + lj * EDIM, msum,
           MROWS, EDIM, FDIM, EDIM, EPI_ACCUM_F32);
    }
    residual_kernel<<<nME, 256, 0, stream>>>(hbuf, ls2 + l * EDIM, msum, 0);
  }

  // ---- final LayerNorm -> d_out (fp32) ----
  ln_rows_kernel<<<MROWS, 256, 0, stream>>>(hbuf, norm_w, norm_b,
                                            (float*)d_out, nullptr, nullptr);
}
